// MultiARLSTM_63522566308281
// MI455X (gfx1250) — matrix operand order repaired and checked by reading, not compiled, not run on hardware
//
#include <hip/hip_runtime.h>

// ---------------------------------------------------------------------------
// Types for CDNA5 WMMA (wave32): v_wmma_f32_16x16x32_bf16
// ---------------------------------------------------------------------------
typedef __bf16 bf16;
typedef __attribute__((ext_vector_type(16))) __bf16 v16bf;
typedef __attribute__((ext_vector_type(8)))  __bf16 v8bf;
typedef __attribute__((ext_vector_type(8)))  float  v8f;
typedef __attribute__((ext_vector_type(4)))  unsigned int u32x4;

#define B_   32
#define T_   2048
#define E_   128
#define H_   512
#define NWG  16           // cooperative workgroups for the LSTM recurrence
#define MROWS (B_ * T_)   // 65536

// ---------------------------------------------------------------------------
// bf16 helpers (bit-level, RNE)
// ---------------------------------------------------------------------------
__device__ __forceinline__ bf16 f2bf(float f) {
  union { float f; unsigned u; } x; x.f = f;
  unsigned r = x.u + 0x7fffu + ((x.u >> 16) & 1u);
  union { unsigned short s; bf16 b; } y; y.s = (unsigned short)(r >> 16);
  return y.b;
}
__device__ __forceinline__ float bfbits2f(unsigned short s) {
  union { unsigned u; float f; } x; x.u = ((unsigned)s) << 16;
  return x.f;
}
__device__ __forceinline__ float bf2f(bf16 b) {
  union { unsigned short s; bf16 b; } y; y.b = b;
  return bfbits2f(y.s);
}
__device__ __forceinline__ float sigmoidf_(float x) {
  return 1.0f / (1.0f + __expf(-x));
}

// ---------------------------------------------------------------------------
// Fragment loads, all 128-bit LDS accesses.
// A (16x32 bf16, row-major LDS, stride lda): lane l holds row M=l&15; its 16
// values are K = hi*16..hi*16+15 -> two contiguous v8bf.
// ---------------------------------------------------------------------------
__device__ __forceinline__ v16bf load_frag_a(const bf16* a, int lda) {
  int lane = threadIdx.x & 31;
  int m = lane & 15, hi = lane >> 4;
  v8bf lo = *(const v8bf*)(a + m * lda + hi * 16);
  v8bf hh = *(const v8bf*)(a + m * lda + hi * 16 + 8);
  return __builtin_shufflevector(lo, hh, 0, 1, 2, 3, 4, 5, 6, 7,
                                 8, 9, 10, 11, 12, 13, 14, 15);
}
// B tile stored in fragment-native LDS layout: 32 lanes x 16 contiguous bf16
// (lane l = hi*16+n holds B[k = hi*16+e][n], e = 0..15).
__device__ __forceinline__ v16bf load_frag_b_packed(const bf16* bfrag) {
  int lane = threadIdx.x & 31;
  v8bf lo = *(const v8bf*)(bfrag + lane * 16);
  v8bf hh = *(const v8bf*)(bfrag + lane * 16 + 8);
  return __builtin_shufflevector(lo, hh, 0, 1, 2, 3, 4, 5, 6, 7,
                                 8, 9, 10, 11, 12, 13, 14, 15);
}
__device__ __forceinline__ v8f wmma_bf16(v16bf a, v16bf b, v8f c) {
  return __builtin_amdgcn_wmma_f32_16x16x32_bf16(false, a, false, b,
                                                 (short)0, c, false, false);
}

// ---------------------------------------------------------------------------
// Conversion / padding kernels (fp32 -> bf16)
// ---------------------------------------------------------------------------
__global__ void cvt_pad_cols(const float* __restrict__ src, bf16* __restrict__ dst,
                             int K, int Kp, long total) {
  long idx = (long)blockIdx.x * blockDim.x + threadIdx.x;
  if (idx >= total) return;
  long r = idx / Kp;
  int  c = (int)(idx - r * Kp);
  dst[idx] = f2bf(c < K ? src[r * K + c] : 0.0f);
}
__global__ void cvt_pad_rows(const float* __restrict__ src, bf16* __restrict__ dst,
                             int rowsSrc, int cols, long total) {
  long idx = (long)blockIdx.x * blockDim.x + threadIdx.x;
  if (idx >= total) return;
  long r = idx / cols;
  int  c = (int)(idx - r * cols);
  dst[idx] = f2bf(r < rowsSrc ? src[r * cols + c] : 0.0f);
}
__global__ void combine_bias(const float* __restrict__ a, const float* __restrict__ b,
                             float* __restrict__ o, int n) {
  int i = blockIdx.x * blockDim.x + threadIdx.x;
  if (i < n) o[i] = a[i] + b[i];
}
__global__ void init_state(bf16* __restrict__ hglob, unsigned* __restrict__ cnt, int n) {
  int i = blockIdx.x * blockDim.x + threadIdx.x;
  if (i < n) hglob[i] = f2bf(0.0f);
  if (i == 0) *cnt = 0u;
}

// ---------------------------------------------------------------------------
// Generic bf16 WMMA GEMM: C[M x N] = act(A[M x K] * B[K x N] + bias)
// 64x64 block tile, 256 threads (8 wave32). K % 32 == 0, M % 64 == 0,
// N % 64 == 0. Wave w -> m-tile (w&3), n-tiles { (w>>2)*2, (w>>2)*2+1 }.
// A staged row-major (b128 in/out); B staged into fragment-native layout
// (coalesced b128 global reads, b16 LDS scatter) so the hot fragment loads
// are pure ds_load_b128.
// ---------------------------------------------------------------------------
__global__ __launch_bounds__(256) void gemm_bf16(
    const bf16* __restrict__ A, int lda,
    const bf16* __restrict__ Bm, int ldb,
    const float* __restrict__ bias,
    bf16* __restrict__ Cb, float* __restrict__ Cf, int ldc,
    int K, int relu) {
  __shared__ bf16 sA[64 * 32];          // row-major
  __shared__ bf16 sB[4 * 32 * 16];      // 4 n-tiles, fragment layout
  const int bm = blockIdx.y * 64;
  const int bn = blockIdx.x * 64;
  const int t  = threadIdx.x;
  const int w  = t >> 5;
  const int wm = w & 3;
  const int wnh = w >> 2;
  const int lane = t & 31;
  const int nl = lane & 15, hi = lane >> 4;

  v8f acc0 = {0, 0, 0, 0, 0, 0, 0, 0};
  v8f acc1 = {0, 0, 0, 0, 0, 0, 0, 0};

  for (int k0 = 0; k0 < K; k0 += 32) {
    {  // A tile: 64 rows x 32 cols, one b128 per thread each way
      int r = t >> 2, c = (t & 3) * 8;
      *(u32x4*)&sA[r * 32 + c] =
          *(const u32x4*)&A[(long)(bm + r) * lda + k0 + c];
    }
    {  // B tile: coalesced b128 read, scatter into fragment layout
      int r = t >> 3, c0 = (t & 7) * 8;
      union { u32x4 v; unsigned short h[8]; } u;
      u.v = *(const u32x4*)&Bm[(long)(k0 + r) * ldb + bn + c0];
      int hi2 = r >> 4, e = r & 15;
#pragma unroll
      for (int i = 0; i < 8; ++i) {
        int c = c0 + i;
        ((unsigned short*)sB)[(((c >> 4) * 32) + hi2 * 16 + (c & 15)) * 16 + e] =
            u.h[i];
      }
    }
    __syncthreads();
    if (k0 + 32 < K)  // pull next A tile toward WGP caches (global_prefetch_b8)
      __builtin_prefetch(&A[(long)(bm + (t >> 2)) * lda + k0 + 32], 0, 0);
    v16bf fa  = load_frag_a(&sA[wm * 16 * 32], 32);
    v16bf fb0 = load_frag_b_packed(&sB[(wnh * 2 + 0) * 512]);
    v16bf fb1 = load_frag_b_packed(&sB[(wnh * 2 + 1) * 512]);
    acc0 = wmma_bf16(fa, fb0, acc0);
    acc1 = wmma_bf16(fa, fb1, acc1);
    __syncthreads();
  }

#pragma unroll
  for (int v = 0; v < 8; ++v) {
    long row = bm + wm * 16 + hi * 8 + v;
    int  c0  = bn + (wnh * 2 + 0) * 16 + nl;
    int  c1  = bn + (wnh * 2 + 1) * 16 + nl;
    float v0 = acc0[v] + (bias ? bias[c0] : 0.0f);
    float v1 = acc1[v] + (bias ? bias[c1] : 0.0f);
    if (relu) { v0 = fmaxf(v0, 0.0f); v1 = fmaxf(v1, 0.0f); }
    if (Cb) { Cb[row * ldc + c0] = f2bf(v0); Cb[row * ldc + c1] = f2bf(v1); }
    if (Cf) { Cf[row * ldc + c0] = v0;       Cf[row * ldc + c1] = v1; }
  }
}

// ---------------------------------------------------------------------------
// Cooperative LSTM recurrence. NWG=16 workgroups x 256 threads.
// WG k owns h-columns [k*32, k*32+32); its 512x128 strided W_h slice lives in
// 128 KB of CDNA5's 320 KB LDS, stored ONCE in fragment-native layout so the
// 2048-step inner loop reads weights purely as ds_load_b128.
// Dynamic LDS: 512*128*2 + 32*512*2 + 32*128*4 = 180224 bytes.
// ---------------------------------------------------------------------------
__global__ __launch_bounds__(256) void lstm_recurrence(
    const bf16* __restrict__ xp,     // (B,T,4H) bf16  (emb @ Wi + bi + bh)
    const bf16* __restrict__ Wh,     // (H,4H)   bf16
    bf16* __restrict__ hglob,        // 2 x (B,H) bf16  double buffer
    bf16* __restrict__ hs,           // (B,T,H)  bf16  all hidden states
    unsigned* __restrict__ barrier_cnt) {
  extern __shared__ char smem[];
  bf16*  sWh = (bf16*)smem;                                   // 512*128 frag layout
  bf16*  sH  = (bf16*)(smem + 512 * 128 * 2);                 // 32*512 row-major
  float* sG  = (float*)(smem + 512 * 128 * 2 + 32 * 512 * 2); // 32*128

  const int wg = blockIdx.x;          // 0..15: h-column slice owner
  const int t  = threadIdx.x;
  const int w  = t >> 5, lane = t & 31;
  const int m  = w & 1;               // batch tile (rows 16m..16m+15)
  const int n0 = (w >> 1) * 2;        // first of two 16-col n-tiles
  const int nl = lane & 15, hi = lane >> 4;

  // One-time W_h slice staging: local col c -> global gate col
  // (c>>5)*512 + wg*32 + (c&31); stored as [k-chunk][n-tile][lane][16].
  for (int s = t * 8; s < 512 * 128; s += 256 * 8) {
    int r = s >> 7, c0 = s & 127;
    union { u32x4 v; unsigned short h[8]; } u;
    u.v = *(const u32x4*)&Wh[(long)r * 2048 + (c0 >> 5) * 512 + wg * 32 + (c0 & 31)];
    int chunk = r >> 5, hi2 = (r >> 4) & 1, e = r & 15;
#pragma unroll
    for (int i = 0; i < 8; ++i) {
      int c = c0 + i;
      ((unsigned short*)sWh)[(((chunk * 8 + (c >> 4)) * 32) + hi2 * 16 + (c & 15)) * 16 + e] =
          u.h[i];
    }
  }

  float creg[4] = {0.0f, 0.0f, 0.0f, 0.0f};  // cell state: 4 (b,j) per thread

  for (int step = 0; step < T_; ++step) {
    const bf16* hsrc = hglob + (size_t)(step & 1) * (B_ * H_);
    bf16*       hdst = hglob + (size_t)((step + 1) & 1) * (B_ * H_);

    for (int idx = t; idx < (B_ * H_) / 8; idx += 256)  // b128 copy
      ((u32x4*)sH)[idx] = ((const u32x4*)hsrc)[idx];
    __syncthreads();

    // Accumulators seeded with xp gate slice (C-matrix of the WMMA chain)
    v8f acc0, acc1;
#pragma unroll
    for (int v = 0; v < 8; ++v) {
      int b   = m * 16 + hi * 8 + v;
      int c0l = (n0 + 0) * 16 + nl;
      int c1l = (n0 + 1) * 16 + nl;
      int g0  = (c0l >> 5) * 512 + wg * 32 + (c0l & 31);
      int g1  = (c1l >> 5) * 512 + wg * 32 + (c1l & 31);
      acc0[v] = bf2f(xp[((long)b * T_ + step) * 2048 + g0]);
      acc1[v] = bf2f(xp[((long)b * T_ + step) * 2048 + g1]);
    }
#pragma unroll
    for (int kk = 0; kk < 512; kk += 32) {
      v16bf fa  = load_frag_a(&sH[(m * 16) * 512 + kk], 512);
      v16bf fb0 = load_frag_b_packed(&sWh[((kk >> 5) * 8 + (n0 + 0)) * 512]);
      v16bf fb1 = load_frag_b_packed(&sWh[((kk >> 5) * 8 + (n0 + 1)) * 512]);
      acc0 = wmma_bf16(fa, fb0, acc0);
      acc1 = wmma_bf16(fa, fb1, acc1);
    }
#pragma unroll
    for (int v = 0; v < 8; ++v) {
      int b = m * 16 + hi * 8 + v;
      sG[b * 128 + (n0 + 0) * 16 + nl] = acc0[v];
      sG[b * 128 + (n0 + 1) * 16 + nl] = acc1[v];
    }
    __syncthreads();

    // Gate nonlinearities; torch order i,f,g,o
#pragma unroll
    for (int p = 0; p < 4; ++p) {
      int lin = t * 4 + p;
      int b = lin >> 5, j = lin & 31;
      float gi = sG[b * 128 +       j];
      float gf = sG[b * 128 +  32 + j];
      float gg = sG[b * 128 +  64 + j];
      float go = sG[b * 128 +  96 + j];
      float cc = sigmoidf_(gf) * creg[p] + sigmoidf_(gi) * tanhf(gg);
      creg[p] = cc;
      float hv = sigmoidf_(go) * tanhf(cc);
      bf16 hb = f2bf(hv);
      hdst[b * 512 + wg * 32 + j] = hb;
      hs[((long)b * T_ + step) * 512 + wg * 32 + j] = hb;
    }
    __syncthreads();

    // Device-wide barrier: CDNA5 cluster barrier (NOP if not clustered) +
    // monotonic atomic counter (no reset races).
    __builtin_amdgcn_s_cluster_barrier();
    if (t == 0) {
      __threadfence();
      atomicAdd(barrier_cnt, 1u);
      unsigned goal = (unsigned)(step + 1) * NWG;
      while (atomicAdd(barrier_cnt, 0u) < goal) __builtin_amdgcn_s_sleep(2);
    }
    __syncthreads();
  }
}

// ---------------------------------------------------------------------------
// Attention logits (E=128 -> A=5) + softmax, one row per thread
// ---------------------------------------------------------------------------
__global__ void attn_softmax_kernel(const bf16* __restrict__ a1,
                                    const float* __restrict__ w2,
                                    const float* __restrict__ b2,
                                    float* __restrict__ attn, long Mrows) {
  long row = (long)blockIdx.x * blockDim.x + threadIdx.x;
  if (row >= Mrows) return;
  float l[5];
#pragma unroll
  for (int a = 0; a < 5; ++a) l[a] = b2[a];
  const u32x4* ap = (const u32x4*)(a1 + row * 128);
#pragma unroll 4
  for (int i = 0; i < 16; ++i) {
    union { u32x4 v; unsigned short h[8]; } u; u.v = ap[i];
#pragma unroll
    for (int j = 0; j < 8; ++j) {
      float v = bfbits2f(u.h[j]);
#pragma unroll
      for (int a = 0; a < 5; ++a) l[a] += v * w2[(i * 8 + j) * 5 + a];
    }
  }
  float mx = l[0];
#pragma unroll
  for (int a = 1; a < 5; ++a) mx = fmaxf(mx, l[a]);
  float s = 0.0f;
#pragma unroll
  for (int a = 0; a < 5; ++a) { l[a] = __expf(l[a] - mx); s += l[a]; }
  float inv = 1.0f / s;
#pragma unroll
  for (int a = 0; a < 5; ++a) attn[row * 5 + a] = l[a] * inv;
}

// ---------------------------------------------------------------------------
// context[b,t,h] = sum_i attn[b,t,i] * hs[b,t-i,h]  (zero-padded)
// ---------------------------------------------------------------------------
__global__ void context_kernel(const float* __restrict__ attn,
                               const bf16* __restrict__ hs,
                               bf16* __restrict__ ctx) {
  long idx = (long)blockIdx.x * blockDim.x + threadIdx.x;  // < MROWS*512
  long row = idx >> 9;
  int  h   = (int)(idx & 511);
  int  tt  = (int)(row & (T_ - 1));
  float s = 0.0f;
#pragma unroll
  for (int i = 0; i < 5; ++i)
    if (tt - i >= 0) s += attn[row * 5 + i] * bf2f(hs[(row - i) * 512 + h]);
  ctx[idx] = f2bf(s);
}

// ---------------------------------------------------------------------------
// dec/ar output heads: 128 -> 1 dot products (b128 activation loads)
// ---------------------------------------------------------------------------
__global__ void heads_kernel(const bf16* __restrict__ d1, const bf16* __restrict__ r1,
                             const float* __restrict__ dw2, const float* __restrict__ db2,
                             const float* __restrict__ aw2, const float* __restrict__ ab2,
                             float* __restrict__ inpart, float* __restrict__ arw,
                             long Mrows) {
  long row = (long)blockIdx.x * blockDim.x + threadIdx.x;
  if (row >= Mrows) return;
  float s1 = 0.0f, s2 = 0.0f;
  const u32x4* dp = (const u32x4*)(d1 + row * 128);
  const u32x4* rp = (const u32x4*)(r1 + row * 128);
#pragma unroll 4
  for (int i = 0; i < 16; ++i) {
    union { u32x4 v; unsigned short h[8]; } ud, ur;
    ud.v = dp[i]; ur.v = rp[i];
#pragma unroll
    for (int j = 0; j < 8; ++j) {
      s1 += bfbits2f(ud.h[j]) * dw2[i * 8 + j];
      s2 += bfbits2f(ur.h[j]) * aw2[i * 8 + j];
    }
  }
  inpart[row] = s1 + db2[0];
  arw[row]    = s2 + ab2[0];
}

// ---------------------------------------------------------------------------
// AR linear recurrence p_t = in_t + ar_t * p_{t-1}, p_{-1}=0.5; out = p*mask
// One lane per batch (32 lanes = exactly one wave32).
// ---------------------------------------------------------------------------
__global__ void ar_scan_kernel(const float* __restrict__ inpart,
                               const float* __restrict__ arw,
                               const float* __restrict__ mask,
                               float* __restrict__ out) {
  int b = threadIdx.x;
  if (b >= B_) return;
  float p = 0.5f;
  for (int tt = 0; tt < T_; ++tt) {
    long r = (long)b * T_ + tt;
    p = inpart[r] + arw[r] * p;
    out[r] = p * mask[r];
  }
}

// ---------------------------------------------------------------------------
// Host orchestration
// ---------------------------------------------------------------------------
extern "C" void kernel_launch(void* const* d_in, const int* in_sizes, int n_in,
                              void* d_out, int out_size, void* d_ws, size_t ws_size,
                              hipStream_t stream) {
  (void)in_sizes; (void)n_in; (void)out_size; (void)ws_size;
  const float* x_a  = (const float*)d_in[0];
  const float* we_a = (const float*)d_in[1];
  const float* be_a = (const float*)d_in[2];
  const float* x_t  = (const float*)d_in[3];
  const float* we_t = (const float*)d_in[4];
  const float* be_t = (const float*)d_in[5];
  const float* x_v  = (const float*)d_in[6];
  const float* we_v = (const float*)d_in[7];
  const float* be_v = (const float*)d_in[8];
  const float* mask = (const float*)d_in[9];
  const float* attn_w1 = (const float*)d_in[10];
  const float* attn_b1 = (const float*)d_in[11];
  const float* attn_w2 = (const float*)d_in[12];
  const float* attn_b2 = (const float*)d_in[13];
  const float* lstm_wi = (const float*)d_in[14];
  const float* lstm_wh = (const float*)d_in[15];
  const float* lstm_bi = (const float*)d_in[16];
  const float* lstm_bh = (const float*)d_in[17];
  const float* dec_w1  = (const float*)d_in[18];
  const float* dec_b1  = (const float*)d_in[19];
  const float* dec_w2  = (const float*)d_in[20];
  const float* dec_b2  = (const float*)d_in[21];
  const float* ar_w1   = (const float*)d_in[22];
  const float* ar_b1   = (const float*)d_in[23];
  const float* ar_w2   = (const float*)d_in[24];
  const float* ar_b2   = (const float*)d_in[25];
  float* out = (float*)d_out;

  // workspace carve-out (deterministic, 256B aligned)
  char* ws = (char*)d_ws;
  size_t off = 0;
  auto alloc = [&](size_t bytes) -> char* {
    char* p = ws + off;
    off = (off + bytes + 255) & ~(size_t)255;
    return p;
  };
  const long M = MROWS;
  bf16* xa_p  = (bf16*)alloc((size_t)M * 96  * 2);
  bf16* xt_p  = (bf16*)alloc((size_t)M * 320 * 2);
  bf16* xv_p  = (bf16*)alloc((size_t)M * 64  * 2);
  bf16* wa_b  = (bf16*)alloc((size_t)96  * 128 * 2);
  bf16* wt_b  = (bf16*)alloc((size_t)320 * 128 * 2);
  bf16* wv_b  = (bf16*)alloc((size_t)64  * 128 * 2);
  bf16* emb   = (bf16*)alloc((size_t)M * 384 * 2);
  bf16* aw1_b = (bf16*)alloc((size_t)384 * 128 * 2);
  bf16* a1    = (bf16*)alloc((size_t)M * 128 * 2);
  float* attn = (float*)alloc((size_t)M * 5 * 4);
  bf16* wi_b  = (bf16*)alloc((size_t)384 * 2048 * 2);
  bf16* wh_b  = (bf16*)alloc((size_t)512 * 2048 * 2);
  float* biasc = (float*)alloc((size_t)2048 * 4);
  bf16* xp    = (bf16*)alloc((size_t)M * 2048 * 2);
  bf16* hglob = (bf16*)alloc((size_t)2 * B_ * H_ * 2);
  bf16* hs    = (bf16*)alloc((size_t)M * 512 * 2);
  bf16* ctx   = (bf16*)alloc((size_t)M * 512 * 2);
  bf16* wd1_b = (bf16*)alloc((size_t)512 * 128 * 2);
  bf16* wr1_b = (bf16*)alloc((size_t)512 * 128 * 2);
  bf16* d1    = (bf16*)alloc((size_t)M * 128 * 2);
  bf16* r1    = (bf16*)alloc((size_t)M * 128 * 2);
  float* inpart = (float*)alloc((size_t)M * 4);
  float* arw    = (float*)alloc((size_t)M * 4);
  unsigned* bcnt = (unsigned*)alloc(256);

  auto blks = [](long total) { return (unsigned)((total + 255) / 256); };

  // 0) init recurrence state + barrier counter
  init_state<<<blks(2 * B_ * H_), 256, 0, stream>>>(hglob, bcnt, 2 * B_ * H_);

  // 1) bf16 conversion / padding
  cvt_pad_cols<<<blks(M *  96), 256, 0, stream>>>(x_a, xa_p,  74,  96, M *  96);
  cvt_pad_cols<<<blks(M * 320), 256, 0, stream>>>(x_t, xt_p, 300, 320, M * 320);
  cvt_pad_cols<<<blks(M *  64), 256, 0, stream>>>(x_v, xv_p,  35,  64, M *  64);
  cvt_pad_rows<<<blks( 96 * 128), 256, 0, stream>>>(we_a, wa_b,  74, 128,  96L * 128);
  cvt_pad_rows<<<blks(320 * 128), 256, 0, stream>>>(we_t, wt_b, 300, 128, 320L * 128);
  cvt_pad_rows<<<blks( 64 * 128), 256, 0, stream>>>(we_v, wv_b,  35, 128,  64L * 128);
  cvt_pad_rows<<<blks(384 * 128), 256, 0, stream>>>(attn_w1, aw1_b, 384, 128, 384L * 128);
  cvt_pad_rows<<<blks(384L * 2048), 256, 0, stream>>>(lstm_wi, wi_b, 384, 2048, 384L * 2048);
  cvt_pad_rows<<<blks(512L * 2048), 256, 0, stream>>>(lstm_wh, wh_b, 512, 2048, 512L * 2048);
  cvt_pad_rows<<<blks(512 * 128), 256, 0, stream>>>(dec_w1, wd1_b, 512, 128, 512L * 128);
  cvt_pad_rows<<<blks(512 * 128), 256, 0, stream>>>(ar_w1,  wr1_b, 512, 128, 512L * 128);
  combine_bias<<<blks(2048), 256, 0, stream>>>(lstm_bi, lstm_bh, biasc, 2048);

  const unsigned gy = (unsigned)(M / 64);  // 1024

  // 2) embeddings -> concatenated emb (B*T, 384), ReLU
  gemm_bf16<<<dim3(2, gy), 256, 0, stream>>>(xa_p,  96, wa_b, 128, be_a,
                                             emb + 0,   nullptr, 384,  96, 1);
  gemm_bf16<<<dim3(2, gy), 256, 0, stream>>>(xt_p, 320, wt_b, 128, be_t,
                                             emb + 128, nullptr, 384, 320, 1);
  gemm_bf16<<<dim3(2, gy), 256, 0, stream>>>(xv_p,  64, wv_b, 128, be_v,
                                             emb + 256, nullptr, 384,  64, 1);

  // 3) attention hidden + softmax over A=5
  gemm_bf16<<<dim3(2, gy), 256, 0, stream>>>(emb, 384, aw1_b, 128, attn_b1,
                                             a1, nullptr, 128, 384, 1);
  attn_softmax_kernel<<<blks(M), 256, 0, stream>>>(a1, attn_w2, attn_b2, attn, M);

  // 4) LSTM input projection xp = emb @ Wi + (bi + bh)
  gemm_bf16<<<dim3(32, gy), 256, 0, stream>>>(emb, 384, wi_b, 2048, biasc,
                                              xp, nullptr, 2048, 384, 0);

  // 5) cooperative LSTM recurrence (16 WGs, 176 KB dynamic LDS each)
  const size_t lstm_lds = 512 * 128 * 2 + 32 * 512 * 2 + 32 * 128 * 4;  // 180224
  lstm_recurrence<<<NWG, 256, lstm_lds, stream>>>(xp, wh_b, hglob, hs, bcnt);

  // 6) local-attention context
  context_kernel<<<blks(M * 512), 256, 0, stream>>>(attn, hs, ctx);

  // 7) dec / ar hidden layers (512 -> 128, ReLU)
  gemm_bf16<<<dim3(2, gy), 256, 0, stream>>>(ctx, 512, wd1_b, 128, dec_b1,
                                             d1, nullptr, 128, 512, 1);
  gemm_bf16<<<dim3(2, gy), 256, 0, stream>>>(ctx, 512, wr1_b, 128, ar_b1,
                                             r1, nullptr, 128, 512, 1);

  // 8) output heads + AR scan
  heads_kernel<<<blks(M), 256, 0, stream>>>(d1, r1, dec_w2, dec_b2, ar_w2, ar_b2,
                                            inpart, arw, M);
  ar_scan_kernel<<<1, 32, 0, stream>>>(inpart, arw, mask, out);
}
